// TitansMemoryModule_34230889349598
// MI455X (gfx1250) — matrix operand order repaired and checked by reading, not compiled, not run on hardware
//
#include <hip/hip_runtime.h>

typedef _Float16 v16h __attribute__((ext_vector_type(16)));
typedef _Float16 h4   __attribute__((ext_vector_type(4)));
typedef float    v8f  __attribute__((ext_vector_type(8)));
typedef float    f4   __attribute__((ext_vector_type(4)));
typedef int      i4   __attribute__((ext_vector_type(4)));

#define D_DIM 128
#define NSLICE 32
#define MOMENTUM 0.9f
#define LR_MEM 0.1f
#define WEIGHT_DECAY 0.01f
#define SURPRISE_THRESHOLD 0.5f
#define FORGETTING_RATE 0.1f

union HFrag { v16h h; i4 q[2]; };

// ---------------- init: scatter map = -1, zero scalars ----------------
__global__ __launch_bounds__(256) void init_kernel(int* map, float* scal, int cap) {
  int i = blockIdx.x * blockDim.x + threadIdx.x;
  if (i < cap) map[i] = -1;
  if (i == 0) { scal[0] = 0.0f; scal[1] = 0.0f; }
}

// ------- per-row L2 normalize f32 -> f16, optional total sumsq -------
__global__ __launch_bounds__(256) void normalize_kernel(const float* __restrict__ src,
                                                        _Float16* __restrict__ dst,
                                                        float* __restrict__ totalsq,
                                                        int rows) {
  int lane = threadIdx.x & 31;
  int wv   = threadIdx.x >> 5;
  int row  = blockIdx.x * 8 + wv;
  if (row >= rows) return;
  f4 v = *(const f4*)(src + (size_t)row * D_DIM + lane * 4);
  float ss = v.x * v.x + v.y * v.y + v.z * v.z + v.w * v.w;
#pragma unroll
  for (int m = 16; m >= 1; m >>= 1) ss += __shfl_xor(ss, m, 32);
  float inv = 1.0f / fmaxf(sqrtf(ss), 1e-8f);
  h4 o;
  o.x = (_Float16)(v.x * inv); o.y = (_Float16)(v.y * inv);
  o.z = (_Float16)(v.z * inv); o.w = (_Float16)(v.w * inv);
  *(h4*)(dst + (size_t)row * D_DIM + lane * 4) = o;
  if (totalsq && lane == 0) atomicAdd(totalsq, ss);
}

// ---- fused GEMM (mem_n @ x_n^T tile) + max over memory rows ----
// A = memory tiles (streamed, ping-pong double buffer, pointer-increment
// prefetch, branch-free steady state); B = x tile (register-resident).
// C layout: lane owns one x-row, the 8 acc VGPRs span mem rows ->
// the max-reduce is pure VALU, no LDS/shuffles.
__device__ __forceinline__ void load_afrag(HFrag* f, const _Float16* base, int hi) {
#pragma unroll
  for (int kk = 0; kk < 4; kk++) {
    int k0 = kk * 32 + hi * 8;
    f[kk].q[0] = *(const i4*)(base + k0);
    f[kk].q[1] = *(const i4*)(base + k0 + 16);
  }
}

__device__ __forceinline__ v8f wmma_tile(const HFrag* a, const v16h* bf) {
  v8f acc = {};
#pragma unroll
  for (int kk = 0; kk < 4; kk++)
    acc = __builtin_amdgcn_wmma_f32_16x16x32_f16(false, a[kk].h, false, bf[kk],
                                                 (short)0, acc, false, false);
  return acc;
}

__global__ __launch_bounds__(256) void simmax_kernel(const _Float16* __restrict__ xh,
                                                     const _Float16* __restrict__ mh,
                                                     float* __restrict__ part,
                                                     int nmt, int nslice) {
  int slice = blockIdx.x % nslice;
  int group = blockIdx.x / nslice;
  int lane  = threadIdx.x & 31;
  int wv    = threadIdx.x >> 5;
  int lm    = lane & 15;
  int hi    = lane >> 4;  // 0 or 1
  int xtile = group * 8 + wv;

  // B fragment: x rows as columns. B 32x16: lane-half K chunks {0-7,16-23}/{8-15,24-31}.
  const _Float16* xrow = xh + ((size_t)xtile * 16 + lm) * D_DIM;
  v16h bf[4];
#pragma unroll
  for (int kk = 0; kk < 4; kk++) {
    HFrag f;
    int k0 = kk * 32 + hi * 8;
    f.q[0] = *(const i4*)(xrow + k0);
    f.q[1] = *(const i4*)(xrow + k0 + 16);
    bf[kk] = f.h;
  }

  // A fragments: memory rows. A 16x32: lane-half K chunks {0-7,16-23}/{8-15,24-31}.
  const _Float16* abase = mh + ((size_t)slice * nmt * 16 + lm) * D_DIM;
  const size_t tile_stride = (size_t)16 * D_DIM;

  HFrag a[4], b[4];
  load_afrag(a, abase, hi);
  load_afrag(b, abase + tile_stride, hi);

  const _Float16* pf = abase + 2 * tile_stride;   // prefetch cursor
  float running = -3.4e38f;

  // steady state: always prefetch, single pointer increment, no clamps
  int steps = (nmt >> 1) - 1;
  for (int it = 0; it < steps; it++) {
    v8f acc0 = wmma_tile(a, bf);
    load_afrag(a, pf, hi);
    v8f acc1 = wmma_tile(b, bf);
    load_afrag(b, pf + tile_stride, hi);
    pf += 2 * tile_stride;

    float r0 = fmaxf(acc0[0], acc0[1]);
    float r1 = fmaxf(acc1[0], acc1[1]);
#pragma unroll
    for (int r = 2; r < 8; r++) { r0 = fmaxf(r0, acc0[r]); r1 = fmaxf(r1, acc1[r]); }
    running = fmaxf(running, fmaxf(r0, r1));
  }

  // epilogue: last two resident tiles, no prefetch
  {
    v8f acc0 = wmma_tile(a, bf);
    v8f acc1 = wmma_tile(b, bf);
    float r0 = fmaxf(acc0[0], acc0[1]);
    float r1 = fmaxf(acc1[0], acc1[1]);
#pragma unroll
    for (int r = 2; r < 8; r++) { r0 = fmaxf(r0, acc0[r]); r1 = fmaxf(r1, acc1[r]); }
    running = fmaxf(running, fmaxf(r0, r1));
  }

  // lo lanes cover mem rows +0..7, hi lanes +8..15 -> two partials per x-row
  part[((size_t)xtile * 16 + lm) * (nslice * 2) + slice * 2 + hi] = running;
}

// -------- reduce partial maxes per batch row -> surprises --------
__global__ void rowmax_surprise_kernel(const float* __restrict__ part,
                                       const float* __restrict__ memsq,
                                       float* __restrict__ out_s,
                                       int nparts) {
  int b = blockIdx.x;
  float m = -3.4e38f;
  for (int i = threadIdx.x; i < nparts; i += blockDim.x)
    m = fmaxf(m, part[(size_t)b * nparts + i]);
#pragma unroll
  for (int k = 16; k >= 1; k >>= 1) m = fmaxf(m, __shfl_xor(m, k, 32));
  __shared__ float sm[8];
  int lane = threadIdx.x & 31, wv = threadIdx.x >> 5;
  int nwv = blockDim.x >> 5;
  if (lane == 0) sm[wv] = m;
  __syncthreads();
  if (threadIdx.x == 0) {
    float mx = sm[0];
    for (int i = 1; i < nwv; i++) mx = fmaxf(mx, sm[i]);
    out_s[b] = (*memsq == 0.0f) ? 1.0f : (1.0f - mx);
  }
}

// ---- sequential pointer scan -> scatter map (indices are distinct) ----
__global__ void scan_kernel(const float* __restrict__ surprises, const int* __restrict__ ptr_in,
                            int* __restrict__ map, int* __restrict__ p_out, int B, int cap) {
  if (threadIdx.x != 0 || blockIdx.x != 0) return;
  long long p = (long long)(*ptr_in);
  for (int b = 0; b < B; b++) {
    bool doit = (surprises[b] > SURPRISE_THRESHOLD) || (p < (long long)cap);
    if (doit) { map[(int)(p % (long long)cap)] = b; p++; }
  }
  *p_out = (int)p;
}

// ----- deterministic two-stage sum of post-scan surprise scores -----
__global__ __launch_bounds__(256) void ss_sum_kernel(const float* __restrict__ ss_in,
                                                     const int* __restrict__ map,
                                                     const float* __restrict__ s,
                                                     float* __restrict__ partial, int cap) {
  int c = blockIdx.x * blockDim.x + threadIdx.x;
  float v = 0.0f;
  if (c < cap) { int b = map[c]; v = (b >= 0) ? s[b] : ss_in[c]; }
#pragma unroll
  for (int m = 16; m >= 1; m >>= 1) v += __shfl_xor(v, m, 32);
  __shared__ float sm[8];
  int lane = threadIdx.x & 31, wv = threadIdx.x >> 5;
  if (lane == 0) sm[wv] = v;
  __syncthreads();
  if (threadIdx.x == 0) {
    float t = 0.0f;
    for (int i = 0; i < 8; i++) t += sm[i];
    partial[blockIdx.x] = t;
  }
}

__global__ __launch_bounds__(256) void sum_partials_kernel(const float* __restrict__ partial,
                                                           float* __restrict__ out, int n) {
  float v = (threadIdx.x < n) ? partial[threadIdx.x] : 0.0f;
#pragma unroll
  for (int m = 16; m >= 1; m >>= 1) v += __shfl_xor(v, m, 32);
  __shared__ float sm[8];
  int lane = threadIdx.x & 31, wv = threadIdx.x >> 5;
  if (lane == 0) sm[wv] = v;
  __syncthreads();
  if (threadIdx.x == 0) {
    float t = 0.0f;
    for (int i = 0; i < 8; i++) t += sm[i];
    *out = t;
  }
}

// ---- fused momentum/memory update + adaptive forgetting + writeout ----
__global__ __launch_bounds__(128) void finalize_kernel(const float* __restrict__ x,
                                                       const float* __restrict__ mem_in,
                                                       const float* __restrict__ mom_in,
                                                       const float* __restrict__ ss_in,
                                                       const int* __restrict__ map,
                                                       const float* __restrict__ s,
                                                       const float* __restrict__ sum_ss,
                                                       const int* __restrict__ p_final,
                                                       float* __restrict__ out_mem,
                                                       float* __restrict__ out_ss,
                                                       float* __restrict__ out_mom, int cap) {
  int c = blockIdx.x;
  int d = threadIdx.x;
  int b = map[c];
  float ssv = (b >= 0) ? s[b] : ss_in[c];
  float avg = (*sum_ss) / (float)cap;
  bool msk = (ssv < avg * FORGETTING_RATE) && (*p_final >= cap);
  float fac = msk ? (1.0f - WEIGHT_DECAY) : 1.0f;
  float mem_v = mem_in[(size_t)c * D_DIM + d];
  float mom_v = mom_in[(size_t)c * D_DIM + d];
  if (b >= 0) {
    float diff = x[(size_t)b * D_DIM + d] - mem_v;
    mom_v = MOMENTUM * mom_v + (1.0f - MOMENTUM) * diff;
    mem_v = mem_v + LR_MEM * mom_v;
  }
  out_mem[(size_t)c * D_DIM + d] = mem_v * fac;
  out_mom[(size_t)c * D_DIM + d] = mom_v * fac;
  if (d == 0) out_ss[c] = ssv * fac;
}

extern "C" void kernel_launch(void* const* d_in, const int* in_sizes, int n_in,
                              void* d_out, int out_size, void* d_ws, size_t ws_size,
                              hipStream_t stream) {
  const float* x      = (const float*)d_in[0];
  const float* mem_in = (const float*)d_in[1];
  const float* mom_in = (const float*)d_in[2];
  const float* ss_in  = (const float*)d_in[3];
  const int*   ptr_in = (const int*)d_in[4];

  int B = in_sizes[0] / D_DIM;   // 2048
  int C = in_sizes[1] / D_DIM;   // 65536

  float* out_s   = (float*)d_out;
  float* out_mem = out_s + B;
  float* out_ss  = out_mem + (size_t)C * D_DIM;
  float* out_mom = out_ss + C;

  // workspace layout
  char* base = (char*)d_ws;
  _Float16* mh = (_Float16*)base;                       // C*128 f16 = 16 MB
  size_t off = (size_t)C * D_DIM * sizeof(_Float16);
  _Float16* xh = (_Float16*)(base + off);               // B*128 f16
  off += (size_t)B * D_DIM * sizeof(_Float16);
  float* part = (float*)(base + off);                   // B * NSLICE*2 f32 (reused for sum partials)
  size_t part_elems = (size_t)B * NSLICE * 2;
  off += part_elems * sizeof(float);
  int* map = (int*)(base + off);                        // C ints
  off += (size_t)C * sizeof(int);
  float* scal = (float*)(base + off);                   // [0]=mem sumsq, [1]=sum_ss, [2]=p_final(int)

  init_kernel<<<(C + 255) / 256, 256, 0, stream>>>(map, scal, C);
  normalize_kernel<<<(C + 7) / 8, 256, 0, stream>>>(mem_in, mh, &scal[0], C);
  normalize_kernel<<<(B + 7) / 8, 256, 0, stream>>>(x, xh, nullptr, B);

  int ngroups = (B / 16) / 8;          // 16 groups of 8 x-tiles
  int nmt = C / (NSLICE * 16);         // 128 memory tiles per slice
  simmax_kernel<<<ngroups * NSLICE, 256, 0, stream>>>(xh, mh, part, nmt, NSLICE);
  rowmax_surprise_kernel<<<B, 64, 0, stream>>>(part, &scal[0], out_s, NSLICE * 2);

  scan_kernel<<<1, 32, 0, stream>>>(out_s, ptr_in, map, (int*)&scal[2], B, C);
  int nblk = (C + 255) / 256;          // 256 partial sums, reuse `part`
  ss_sum_kernel<<<nblk, 256, 0, stream>>>(ss_in, map, out_s, part, C);
  sum_partials_kernel<<<1, 256, 0, stream>>>(part, &scal[1], nblk);
  finalize_kernel<<<C, 128, 0, stream>>>(x, mem_in, mom_in, ss_in, map, out_s,
                                         &scal[1], (int*)&scal[2],
                                         out_mem, out_ss, out_mom, C);
}
